// SubInfoGraph_1151051235811
// MI455X (gfx1250) — matrix operand order repaired and checked by reading, not compiled, not run on hardware
//
#include <hip/hip_runtime.h>
#include <math.h>

// ---------------------------------------------------------------------------
// InfoGraph forward (GIN x5 + local/global MLP heads + JSD MI loss + prior)
// for gfx1250 (MI455X): bf16 WMMA (v_wmma_f32_16x16x32_bf16) for all GEMMs,
// f32 accumulate; edge aggregation via L2-resident global_atomic_add_f32;
// res-matrix softplus reduction fused into the WMMA epilogue.
// Weights are pre-transposed+converted to bf16 [N x K] so each B operand is
// one contiguous 32B v16bf load per K-step (no strided dword loads, no cvt).
// Each wave computes a 16x32 output strip: one A tile x two B tiles.
// ---------------------------------------------------------------------------

typedef __attribute__((ext_vector_type(16))) __bf16 v16bf;
typedef __attribute__((ext_vector_type(8)))  float  v8f;

#define NNODES   100000
#define NEDGES   1600000
#define NFEAT    128
#define GDIM     64
#define EMB      320
#define NGRAPHS  512
#define LOG2C    0.6931471805599453f

__device__ __forceinline__ float softplusf(float x) {
  // stable: max(x,0) + log(1 + exp(-|x|)); fast transcendental path
  // (v_exp_f32 + v_log_f32) -- precision amply covered by bf16 matmul noise.
  return fmaxf(x, 0.f) + __logf(1.f + __expf(-fabsf(x)));
}

// ---------------- WT[n*K + k] = bf16(W[k*N + n])  (weight transpose) -------
__global__ __launch_bounds__(256) void k_wt(const float* __restrict__ W,
                                            __bf16* __restrict__ WT,
                                            int K, int N) {
  int idx = blockIdx.x * blockDim.x + threadIdx.x;
  int total = K * N;
  if (idx >= total) return;
  int n = idx / K;
  int k = idx - n * K;
  WT[idx] = (__bf16)W[(size_t)k * N + n];
}

// ---------------- Xb[t] = bf16(X[t])  (plain convert) ----------------------
__global__ __launch_bounds__(256) void k_cvt(const float* __restrict__ X,
                                             __bf16* __restrict__ Xb,
                                             long long total) {
  long long t = (long long)blockIdx.x * blockDim.x + threadIdx.x;
  if (t >= total) return;
  Xb[t] = (__bf16)X[t];
}

// ------------------------- z = x (strided copy) ----------------------------
__global__ __launch_bounds__(256) void k_copy(const float* __restrict__ x, int ldx,
                                              float* __restrict__ z, int ldz,
                                              int logc, long long total) {
  long long t = (long long)blockIdx.x * blockDim.x + threadIdx.x;
  long long stride = (long long)gridDim.x * blockDim.x;
  int mask = (1 << logc) - 1;
  for (; t < total; t += stride) {
    int row = (int)(t >> logc);
    int c = (int)t & mask;
    z[(size_t)row * ldz + c] = x[(size_t)row * ldx + c];
  }
}

// ------------------- z[dst] += x[src] over all edges -----------------------
__global__ __launch_bounds__(256) void k_edge_scatter(const float* __restrict__ x, int ldx,
                                                      const int* __restrict__ src,
                                                      const int* __restrict__ dst,
                                                      float* __restrict__ z, int ldz,
                                                      int logc, long long total) {
  long long t = (long long)blockIdx.x * blockDim.x + threadIdx.x;
  long long stride = (long long)gridDim.x * blockDim.x;
  int mask = (1 << logc) - 1;
  for (; t < total; t += stride) {
    long long e = t >> logc;
    int c = (int)t & mask;
    int s = src[e], d = dst[e];
    atomicAdd(&z[(size_t)d * ldz + c], x[(size_t)s * ldx + c]);
  }
}

// ---------------------------------------------------------------------------
// D[M x N] = act(A[M x K] @ W[K x N] + bias[N]) (+ S[M x N] if ADDS)
// WT is the bf16 transposed weight [N x K].  ACT: 0 none, 1 relu, 2 sigmoid.
// One wave per 16x32 strip (two 16x16 tiles); K step 32.
// bf16 WMMA operand layouts as loaded here (A, B share k order):
//  A 16x32: lane L row=L&15; elem j: K = (L>>4)*16 + j
//  B 32x16: lane L col=L&15; elem j: K = (L>>4)*16 + j  -> contiguous in WT
//  C 16x16: lane L col=L&15; vgpr r: M = r + (L>>4)*8
// ---------------------------------------------------------------------------
template <int ACT, int ADDS>
__global__ __launch_bounds__(256) void k_wmma_gemm(const float* __restrict__ A, int lda,
                                                   const __bf16* __restrict__ WT,
                                                   const float* __restrict__ bias,
                                                   const float* __restrict__ S,
                                                   float* __restrict__ D, int ldd,
                                                   int Mrows, int N, int K) {
  int lane = threadIdx.x & 31;
  int wave = (int)(((long long)blockIdx.x * blockDim.x + threadIdx.x) >> 5);
  int nWaves = (int)(((long long)gridDim.x * blockDim.x) >> 5);
  int stripsN = N >> 5;                       // pairs of 16-col tiles
  int totalStrips = (Mrows >> 4) * stripsN;
  int hh = lane >> 4;
  int l15 = lane & 15;
  for (int strip = wave; strip < totalStrips; strip += nWaves) {
    int tm = strip / stripsN;
    int tn = strip - tm * stripsN;
    int row0 = tm << 4, col0 = tn << 5;
    v8f acc0 = {0.f, 0.f, 0.f, 0.f, 0.f, 0.f, 0.f, 0.f};
    v8f acc1 = {0.f, 0.f, 0.f, 0.f, 0.f, 0.f, 0.f, 0.f};
    const float* Arow = A + (size_t)(row0 + l15) * lda;
    int bcol0 = col0 + l15;
    int bcol1 = col0 + 16 + l15;
    const __bf16* Wt0 = WT + (size_t)bcol0 * K;
    const __bf16* Wt1 = WT + (size_t)bcol1 * K;
#pragma unroll 2
    for (int k0 = 0; k0 < K; k0 += 32) {
      v16bf a;
      int abase = k0 + hh * 16;
#pragma unroll
      for (int j = 0; j < 8; ++j) {
        a[j]     = (__bf16)Arow[abase + j];
        a[j + 8] = (__bf16)Arow[abase + 8 + j];
      }
      int kb = k0 + hh * 16;
      v16bf b0 = *(const v16bf*)(Wt0 + kb);   // 32B contiguous, K-major
      v16bf b1 = *(const v16bf*)(Wt1 + kb);
      acc0 = __builtin_amdgcn_wmma_f32_16x16x32_bf16(false, a, false, b0,
                                                     (short)0, acc0, false, false);
      acc1 = __builtin_amdgcn_wmma_f32_16x16x32_bf16(false, a, false, b1,
                                                     (short)0, acc1, false, false);
    }
    float bv0 = bias[bcol0];
    float bv1 = bias[bcol1];
#pragma unroll
    for (int r = 0; r < 8; ++r) {
      int m = row0 + hh * 8 + r;
      float v0 = acc0[r] + bv0;
      float v1 = acc1[r] + bv1;
      if (ACT == 1) { v0 = fmaxf(v0, 0.f); v1 = fmaxf(v1, 0.f); }
      else if (ACT == 2) {
        v0 = __builtin_amdgcn_rcpf(1.f + __expf(-v0));
        v1 = __builtin_amdgcn_rcpf(1.f + __expf(-v1));
      }
      if (ADDS) {
        v0 += S[(size_t)m * ldd + bcol0];
        v1 += S[(size_t)m * ldd + bcol1];
      }
      D[(size_t)m * ldd + bcol0] = v0;
      D[(size_t)m * ldd + bcol1] = v1;
    }
  }
}

// ---------------- BN statistics: column sums / sumsq over rows -------------
__global__ __launch_bounds__(256) void k_bn_stats(const float* __restrict__ h,
                                                  float* __restrict__ sum,
                                                  float* __restrict__ sumsq, int nRows) {
  __shared__ float sS[4][64];
  __shared__ float sQ[4][64];
  int c = threadIdx.x & 63;
  int r = threadIdx.x >> 6;  // 0..3
  int row0 = blockIdx.x * 256;
  float s = 0.f, q = 0.f;
  for (int i = r; i < 256; i += 4) {
    int row = row0 + i;
    if (row < nRows) {
      float v = h[(size_t)row * 64 + c];
      s += v;
      q += v * v;
    }
  }
  sS[r][c] = s;
  sQ[r][c] = q;
  __syncthreads();
  if (r == 0) {
    s = sS[0][c] + sS[1][c] + sS[2][c] + sS[3][c];
    q = sQ[0][c] + sQ[1][c] + sQ[2][c] + sQ[3][c];
    atomicAdd(&sum[c], s);
    atomicAdd(&sumsq[c], q);
  }
}

// ---- BN apply -> write into M[:, col0..col0+64) and pool into Y -----------
__global__ __launch_bounds__(256) void k_bn_apply_pool(const float* __restrict__ h,
                                                       const float* __restrict__ sum,
                                                       const float* __restrict__ sumsq,
                                                       const float* __restrict__ gam,
                                                       const float* __restrict__ bet,
                                                       const int* __restrict__ gidx,
                                                       float* __restrict__ M,
                                                       float* __restrict__ Y,
                                                       int col0, int nRows) {
  long long t = (long long)blockIdx.x * blockDim.x + threadIdx.x;
  long long total = (long long)nRows * 64;
  long long stride = (long long)gridDim.x * blockDim.x;
  const float invN = 1.f / (float)NNODES;
  for (; t < total; t += stride) {
    int row = (int)(t >> 6);
    int c = (int)t & 63;
    float mu = sum[c] * invN;
    float var = sumsq[c] * invN - mu * mu;
    float v = (h[t] - mu) * rsqrtf(var + 1e-5f) * gam[c] + bet[c];
    M[(size_t)row * EMB + col0 + c] = v;
    atomicAdd(&Y[(size_t)gidx[row] * EMB + col0 + c], v);
  }
}

// ---------------------------------------------------------------------------
// Fused res = l_enc @ g_enc^T with JSD softplus reduction (no materialized
// [100K x 512] matrix). g_enc pre-converted to bf16 (row layout == B layout).
// Pos entries: LOG2 - sp(-r); neg: sp(r) - LOG2. Two graph tiles per wave.
// ---------------------------------------------------------------------------
__global__ __launch_bounds__(256) void k_res_loss(const float* __restrict__ L,
                                                  const __bf16* __restrict__ Gb,
                                                  const int* __restrict__ gidx,
                                                  float* __restrict__ accP,
                                                  float* __restrict__ accN) {
  int lane = threadIdx.x & 31;
  int wave = (int)(((long long)blockIdx.x * blockDim.x + threadIdx.x) >> 5);
  int nWaves = (int)(((long long)gridDim.x * blockDim.x) >> 5);
  const int stripsG = NGRAPHS / 32;                 // 16
  const int totalStrips = (NNODES / 16) * stripsG;  // 100000
  int hh = lane >> 4;
  int l15 = lane & 15;
  float aP = 0.f, aN = 0.f;
  for (int strip = wave; strip < totalStrips; strip += nWaves) {
    int tm = strip / stripsG;
    int tg = strip - tm * stripsG;
    int row0 = tm << 4, g0 = tg << 5;
    v8f acc0 = {0.f, 0.f, 0.f, 0.f, 0.f, 0.f, 0.f, 0.f};
    v8f acc1 = {0.f, 0.f, 0.f, 0.f, 0.f, 0.f, 0.f, 0.f};
    const float* Lrow = L + (size_t)(row0 + l15) * EMB;
    const __bf16* Grow0 = Gb + (size_t)(g0 + l15) * EMB;       // B col = graph
    const __bf16* Grow1 = Gb + (size_t)(g0 + 16 + l15) * EMB;
#pragma unroll 2
    for (int k0 = 0; k0 < EMB; k0 += 32) {
      v16bf a;
      int abase = k0 + hh * 16;
#pragma unroll
      for (int j = 0; j < 8; ++j) {
        a[j]     = (__bf16)Lrow[abase + j];
        a[j + 8] = (__bf16)Lrow[abase + 8 + j];
      }
      int kb = k0 + hh * 16;
      v16bf b0 = *(const v16bf*)(Grow0 + kb);
      v16bf b1 = *(const v16bf*)(Grow1 + kb);
      acc0 = __builtin_amdgcn_wmma_f32_16x16x32_bf16(false, a, false, b0,
                                                     (short)0, acc0, false, false);
      acc1 = __builtin_amdgcn_wmma_f32_16x16x32_bf16(false, a, false, b1,
                                                     (short)0, acc1, false, false);
    }
    int gcol0 = g0 + l15;
    int gcol1 = g0 + 16 + l15;
#pragma unroll
    for (int r = 0; r < 8; ++r) {
      int node = row0 + hh * 8 + r;
      int gi = gidx[node];
      float r0 = acc0[r];
      float r1 = acc1[r];
      if (gi == gcol0) aP += (LOG2C - softplusf(-r0));
      else             aN += (softplusf(r0) - LOG2C);
      if (gi == gcol1) aP += (LOG2C - softplusf(-r1));
      else             aN += (softplusf(r1) - LOG2C);
    }
  }
  for (int off = 16; off > 0; off >>= 1) {
    aP += __shfl_down(aP, off, 32);
    aN += __shfl_down(aN, off, 32);
  }
  __shared__ float sP[8];
  __shared__ float sN[8];
  int w = threadIdx.x >> 5;
  if (lane == 0) { sP[w] = aP; sN[w] = aN; }
  __syncthreads();
  if (threadIdx.x == 0) {
    float p = 0.f, n = 0.f;
    for (int i = 0; i < 8; ++i) { p += sP[i]; n += sN[i]; }
    atomicAdd(accP, p);
    atomicAdd(accN, n);
  }
}

// ------------ prior head layer 2: s = t_row . w2 + b2; reduce log ----------
// mode 0: acc += log(sigmoid(s)) = -sp(-s);  mode 1: acc += log(1-sig(s)) = -sp(s)
__global__ __launch_bounds__(256) void k_prior_reduce(const float* __restrict__ T,
                                                      const float* __restrict__ w2,
                                                      const float* __restrict__ b2,
                                                      int mode, float* __restrict__ acc) {
  int lane = threadIdx.x & 31;
  int wave = (int)(((long long)blockIdx.x * blockDim.x + threadIdx.x) >> 5);
  if (wave >= NGRAPHS) return;
  const float* row = T + (size_t)wave * EMB;
  float s = 0.f;
  for (int k = lane; k < EMB; k += 32) s += row[k] * w2[k];
  for (int off = 16; off > 0; off >>= 1) s += __shfl_down(s, off, 32);
  if (lane == 0) {
    s += b2[0];
    float val = mode ? -softplusf(s) : -softplusf(-s);
    atomicAdd(acc, val);
  }
}

__global__ void k_combine(const float* __restrict__ sc, float* __restrict__ out) {
  float E_pos = sc[0] / (float)NNODES;
  float E_neg = sc[1] / ((float)NNODES * (float)(NGRAPHS - 1));
  float prior = -0.1f * (sc[2] / (float)NGRAPHS + sc[3] / (float)NGRAPHS);
  out[0] = (E_neg - E_pos) + prior;
}

// ---------------------------------------------------------------------------
struct Ptrs {
  const float *nodef, *noise;
  const float *gl1w[5], *gl1b[5], *gl2w[5], *gl2b[5], *bng[5], *bnb[5];
  const float *lo_l1w, *lo_l1b, *lo_l2w, *lo_l2b, *lo_skw, *lo_skb;
  const float *g_l1w, *g_l1b, *g_l2w, *g_l2b, *g_skw, *g_skb;
  const float *pr_l1w, *pr_l1b, *pr_l2w, *pr_l2b;
  const int *src, *dst, *gidx;
};

extern "C" void kernel_launch(void* const* d_in, const int* in_sizes, int n_in,
                              void* d_out, int out_size, void* d_ws, size_t ws_size,
                              hipStream_t stream) {
  Ptrs P;
  auto Fp = [&](int i) { return (const float*)d_in[i]; };
  auto Ip = [&](int i) { return (const int*)d_in[i]; };

  if (in_sizes[0] == NNODES * NFEAT) {
    // Insertion-order flattening of setup_inputs():
    // node_features, prior_noise, params{gin[5]{l1.w,l1.b,l2.w,l2.b,bn_g,bn_b},
    // local{w,b,w,b,skw,skb}, global{...}, prior{l1w,l1b,l2w,l2b}}, edge, gidx
    P.nodef = Fp(0); P.noise = Fp(1);
    for (int i = 0; i < 5; ++i) {
      int b = 2 + 6 * i;
      P.gl1w[i] = Fp(b + 0); P.gl1b[i] = Fp(b + 1);
      P.gl2w[i] = Fp(b + 2); P.gl2b[i] = Fp(b + 3);
      P.bng[i]  = Fp(b + 4); P.bnb[i]  = Fp(b + 5);
    }
    P.lo_l1w = Fp(32); P.lo_l1b = Fp(33); P.lo_l2w = Fp(34);
    P.lo_l2b = Fp(35); P.lo_skw = Fp(36); P.lo_skb = Fp(37);
    P.g_l1w = Fp(38); P.g_l1b = Fp(39); P.g_l2w = Fp(40);
    P.g_l2b = Fp(41); P.g_skw = Fp(42); P.g_skb = Fp(43);
    P.pr_l1w = Fp(44); P.pr_l1b = Fp(45); P.pr_l2w = Fp(46); P.pr_l2b = Fp(47);
    P.src = Ip(48); P.dst = Ip(48) + NEDGES; P.gidx = Ip(49);
  } else {
    // jax tree_leaves (sorted dict keys): edge_index, graph_index,
    // node_features, params{gin[i]{bn_b,bn_g,l1.b,l1.w,l2.b,l2.w},
    // global{l1.b,l1.w,l2.b,l2.w,skip.b,skip.w}, local{...},
    // prior{l1.b,l1.w,l2.b,l2.w}}, prior_noise
    P.src = Ip(0); P.dst = Ip(0) + NEDGES; P.gidx = Ip(1);
    P.nodef = Fp(2);
    for (int i = 0; i < 5; ++i) {
      int b = 3 + 6 * i;
      P.bnb[i]  = Fp(b + 0); P.bng[i]  = Fp(b + 1);
      P.gl1b[i] = Fp(b + 2); P.gl1w[i] = Fp(b + 3);
      P.gl2b[i] = Fp(b + 4); P.gl2w[i] = Fp(b + 5);
    }
    P.g_l1b = Fp(33); P.g_l1w = Fp(34); P.g_l2b = Fp(35);
    P.g_l2w = Fp(36); P.g_skb = Fp(37); P.g_skw = Fp(38);
    P.lo_l1b = Fp(39); P.lo_l1w = Fp(40); P.lo_l2b = Fp(41);
    P.lo_l2w = Fp(42); P.lo_skb = Fp(43); P.lo_skw = Fp(44);
    P.pr_l1b = Fp(45); P.pr_l1w = Fp(46); P.pr_l2b = Fp(47); P.pr_l2w = Fp(48);
    P.noise = Fp(49);
  }

  // ---- workspace layout (f32 elements, then 32B-aligned bf16 arena) -------
  float* ws = (float*)d_ws;
  size_t off = 0;
  float* Z   = ws + off; off += (size_t)NNODES * 128;   // GIN input+agg
  float* H1  = ws + off; off += (size_t)NNODES * 64;    // hidden of GIN MLP
  float* H   = ws + off; off += (size_t)NNODES * 64;    // pre-BN output
  float* Mm  = ws + off; off += (size_t)NNODES * EMB;   // concat features
  float* T1  = ws + off; off += (size_t)NNODES * EMB;   // local l1 act
  float* LSK = ws + off; off += (size_t)NNODES * EMB;   // skip -> l_enc
  float* Y   = ws + off; off += (size_t)NGRAPHS * EMB;  // pooled graph emb
  float* TG1 = ws + off; off += (size_t)NGRAPHS * EMB;
  float* GSK = ws + off; off += (size_t)NGRAPHS * EMB;  // skip -> g_enc
  float* TP  = ws + off; off += (size_t)NGRAPHS * EMB;
  float* TY  = ws + off; off += (size_t)NGRAPHS * EMB;
  float* bnS = ws + off; off += 64;
  float* bnQ = ws + off; off += 64;
  float* SC  = ws + off; off += 8;  // accP, accN, accA, accB (+pad)
  off = (off + 7) & ~(size_t)7;     // 32B align for bf16 arena

  __bf16* bfa = (__bf16*)(ws + off);
  size_t boff = 0;
  __bf16* WTg1[5]; __bf16* WTg2[5];
  for (int i = 0; i < 5; ++i) { WTg1[i] = bfa + boff; boff += 64 * 128; }
  for (int i = 0; i < 5; ++i) { WTg2[i] = bfa + boff; boff += 64 * 64; }
  __bf16* WTlo1 = bfa + boff; boff += (size_t)EMB * EMB;
  __bf16* WTlo2 = bfa + boff; boff += (size_t)EMB * EMB;
  __bf16* WTlos = bfa + boff; boff += (size_t)EMB * EMB;
  __bf16* WTgl1 = bfa + boff; boff += (size_t)EMB * EMB;
  __bf16* WTgl2 = bfa + boff; boff += (size_t)EMB * EMB;
  __bf16* WTgls = bfa + boff; boff += (size_t)EMB * EMB;
  __bf16* WTpr1 = bfa + boff; boff += (size_t)EMB * EMB;
  __bf16* Gbf   = bfa + boff; boff += (size_t)NGRAPHS * EMB;

  (void)hipMemsetAsync(Y, 0, (size_t)NGRAPHS * EMB * sizeof(float), stream);
  (void)hipMemsetAsync(SC, 0, 4 * sizeof(float), stream);

  // ---- transpose+convert all weights to bf16 [N x K] ----------------------
  for (int i = 0; i < 5; ++i) {
    int K1 = i ? GDIM : NFEAT;
    k_wt<<<(K1 * GDIM + 255) / 256, 256, 0, stream>>>(P.gl1w[i], WTg1[i], K1, GDIM);
    k_wt<<<(GDIM * GDIM + 255) / 256, 256, 0, stream>>>(P.gl2w[i], WTg2[i], GDIM, GDIM);
  }
  int wblk = (EMB * EMB + 255) / 256;
  k_wt<<<wblk, 256, 0, stream>>>(P.lo_l1w, WTlo1, EMB, EMB);
  k_wt<<<wblk, 256, 0, stream>>>(P.lo_l2w, WTlo2, EMB, EMB);
  k_wt<<<wblk, 256, 0, stream>>>(P.lo_skw, WTlos, EMB, EMB);
  k_wt<<<wblk, 256, 0, stream>>>(P.g_l1w, WTgl1, EMB, EMB);
  k_wt<<<wblk, 256, 0, stream>>>(P.g_l2w, WTgl2, EMB, EMB);
  k_wt<<<wblk, 256, 0, stream>>>(P.g_skw, WTgls, EMB, EMB);
  k_wt<<<wblk, 256, 0, stream>>>(P.pr_l1w, WTpr1, EMB, EMB);

  // ---- GIN layers ----------------------------------------------------------
  for (int l = 0; l < 5; ++l) {
    int din = l ? GDIM : NFEAT;
    int logc = l ? 6 : 7;
    const float* X = l ? (Mm + (size_t)(l - 1) * GDIM) : P.nodef;
    int ldx = l ? EMB : NFEAT;

    long long totC = (long long)NNODES * din;
    k_copy<<<(unsigned)((totC + 255) / 256), 256, 0, stream>>>(X, ldx, Z, din, logc, totC);
    long long totE = (long long)NEDGES * din;
    k_edge_scatter<<<(unsigned)((totE + 255) / 256), 256, 0, stream>>>(
        X, ldx, P.src, P.dst, Z, din, logc, totE);

    int strips = (NNODES / 16) * (GDIM / 32);  // 12500
    int blocks = (strips + 7) / 8;
    k_wmma_gemm<1, 0><<<blocks, 256, 0, stream>>>(Z, din, WTg1[l], P.gl1b[l], nullptr,
                                                  H1, GDIM, NNODES, GDIM, din);
    k_wmma_gemm<1, 0><<<blocks, 256, 0, stream>>>(H1, GDIM, WTg2[l], P.gl2b[l], nullptr,
                                                  H, GDIM, NNODES, GDIM, GDIM);

    (void)hipMemsetAsync(bnS, 0, 64 * sizeof(float), stream);
    (void)hipMemsetAsync(bnQ, 0, 64 * sizeof(float), stream);
    k_bn_stats<<<(NNODES + 255) / 256, 256, 0, stream>>>(H, bnS, bnQ, NNODES);
    k_bn_apply_pool<<<(unsigned)(((long long)NNODES * 64 + 255) / 256), 256, 0, stream>>>(
        H, bnS, bnQ, P.bng[l], P.bnb[l], P.gidx, Mm, Y, l * GDIM, NNODES);
  }

  // ---- global head: g_enc = relu(relu(y@W1+b)@W2+b2) + y@Wsk+bsk ----------
  {
    int blocks = (((NGRAPHS / 16) * (EMB / 32)) + 7) / 8;  // 40
    k_wmma_gemm<1, 0><<<blocks, 256, 0, stream>>>(Y, EMB, WTgl1, P.g_l1b, nullptr,
                                                  TG1, EMB, NGRAPHS, EMB, EMB);
    k_wmma_gemm<0, 0><<<blocks, 256, 0, stream>>>(Y, EMB, WTgls, P.g_skb, nullptr,
                                                  GSK, EMB, NGRAPHS, EMB, EMB);
    k_wmma_gemm<1, 1><<<blocks, 256, 0, stream>>>(TG1, EMB, WTgl2, P.g_l2b, GSK,
                                                  GSK, EMB, NGRAPHS, EMB, EMB);
    k_cvt<<<((NGRAPHS * EMB) + 255) / 256, 256, 0, stream>>>(GSK, Gbf,
                                                             (long long)NGRAPHS * EMB);
  }
  // ---- local head on M ----------------------------------------------------
  {
    int blocks = (((NNODES / 16) * (EMB / 32)) + 7) / 8;  // 7813
    k_wmma_gemm<1, 0><<<blocks, 256, 0, stream>>>(Mm, EMB, WTlo1, P.lo_l1b, nullptr,
                                                  T1, EMB, NNODES, EMB, EMB);
    k_wmma_gemm<0, 0><<<blocks, 256, 0, stream>>>(Mm, EMB, WTlos, P.lo_skb, nullptr,
                                                  LSK, EMB, NNODES, EMB, EMB);
    k_wmma_gemm<1, 1><<<blocks, 256, 0, stream>>>(T1, EMB, WTlo2, P.lo_l2b, LSK,
                                                  LSK, EMB, NNODES, EMB, EMB);
  }

  // ---- fused res + JSD loss reduction -------------------------------------
  k_res_loss<<<4096, 256, 0, stream>>>(LSK, Gbf, P.gidx, SC + 0, SC + 1);

  // ---- prior head ---------------------------------------------------------
  {
    int blocks = (((NGRAPHS / 16) * (EMB / 32)) + 7) / 8;
    k_wmma_gemm<2, 0><<<blocks, 256, 0, stream>>>(P.noise, EMB, WTpr1, P.pr_l1b, nullptr,
                                                  TP, EMB, NGRAPHS, EMB, EMB);
    k_wmma_gemm<2, 0><<<blocks, 256, 0, stream>>>(Y, EMB, WTpr1, P.pr_l1b, nullptr,
                                                  TY, EMB, NGRAPHS, EMB, EMB);
    k_prior_reduce<<<(NGRAPHS * 32 + 255) / 256, 256, 0, stream>>>(TP, P.pr_l2w, P.pr_l2b, 0, SC + 2);
    k_prior_reduce<<<(NGRAPHS * 32 + 255) / 256, 256, 0, stream>>>(TY, P.pr_l2w, P.pr_l2b, 1, SC + 3);
  }

  k_combine<<<1, 1, 0, stream>>>(SC, (float*)d_out);
  (void)n_in; (void)out_size; (void)ws_size; (void)in_sizes;
}
